// GraphDiffNet_24000277250259
// MI455X (gfx1250) — matrix operand order repaired and checked by reading, not compiled, not run on hardware
//
#include <hip/hip_runtime.h>
#include <hip/hip_bf16.h>

typedef __bf16 bf16;
typedef __attribute__((ext_vector_type(16))) __bf16 v16bf;
typedef __attribute__((ext_vector_type(8)))  __bf16 v8bfx;
typedef __attribute__((ext_vector_type(8)))  float  v8f;
typedef __attribute__((ext_vector_type(4)))  unsigned int u32x4;
typedef __attribute__((ext_vector_type(8)))  int i32x8;
typedef __attribute__((ext_vector_type(4)))  int i32x4;

#define V_N   6000
#define N_G   6000
#define KEV   128
#define E_N   96000

__device__ __forceinline__ float cvtf(float x) { return x; }
__device__ __forceinline__ float cvtf(bf16 x)  { return (float)x; }

__device__ __forceinline__ v16bf frag_cat(v8bfx lo, v8bfx hi) {
  return __builtin_shufflevector(lo, hi, 0, 1, 2, 3, 4, 5, 6, 7,
                                 8, 9, 10, 11, 12, 13, 14, 15);
}

// 16 contiguous elements -> two v8bfx packs (unguarded fast path)
__device__ __forceinline__ void load16(const float* gp, v8bfx& p0, v8bfx& p1) {
  const float4* q = (const float4*)gp;
  float4 f0 = q[0], f1 = q[1], f2 = q[2], f3 = q[3];
  v8bfx a, b;
  a[0] = (bf16)f0.x; a[1] = (bf16)f0.y; a[2] = (bf16)f0.z; a[3] = (bf16)f0.w;
  a[4] = (bf16)f1.x; a[5] = (bf16)f1.y; a[6] = (bf16)f1.z; a[7] = (bf16)f1.w;
  b[0] = (bf16)f2.x; b[1] = (bf16)f2.y; b[2] = (bf16)f2.z; b[3] = (bf16)f2.w;
  b[4] = (bf16)f3.x; b[5] = (bf16)f3.y; b[6] = (bf16)f3.z; b[7] = (bf16)f3.w;
  p0 = a; p1 = b;
}
__device__ __forceinline__ void load16(const bf16* gp, v8bfx& p0, v8bfx& p1) {
  const v8bfx* q = (const v8bfx*)gp;
  p0 = q[0]; p1 = q[1];
}

// WMMA compute micro-step: preload A frag + all 4 B frags (10x ds_load_b128
// in flight, partial dscnt waits), then 4 back-to-back WMMAs.
__device__ __forceinline__ void wmma_step(const bf16 (*sA)[40], const bf16 (*sBt)[40],
                                          int wave, int mA, int kbA, int kbB, int colB,
                                          v8f acc[4]) {
  const bf16* arow = &sA[wave * 16 + mA][0];
  v8bfx a0 = *(const v8bfx*)(arow + kbA);
  v8bfx a1 = *(const v8bfx*)(arow + kbA + 16);
  v8bfx b0[4], b1[4];
  #pragma unroll
  for (int nt = 0; nt < 4; ++nt) {
    const bf16* bcol = &sBt[nt * 16 + colB][0];
    b0[nt] = *(const v8bfx*)(bcol + kbB);
    b1[nt] = *(const v8bfx*)(bcol + kbB + 16);
  }
  v16bf af = frag_cat(a0, a1);
  #pragma unroll
  for (int nt = 0; nt < 4; ++nt) {
    acc[nt] = __builtin_amdgcn_wmma_f32_16x16x32_bf16(
        false, af, false, frag_cat(b0[nt], b1[nt]), (short)0, acc[nt],
        false, false);
  }
}

// ---------------------------------------------------------------------------
// N=64 GEMM:  C(M x 64) = act( A(M x K) @ B(K x 64) [+ A2@B2] + bias [+ resid] )
//   TRANSA=1 : A stored (K x M)  (only used for the tiny evecs^T split-K op)
//   DUAL=1   : second accumulation pass A2@B2 (optionally negated B2)
//   gridDim.y>1 : split-K, f32 atomic accumulation into pre-zeroed C
// BM=128, BN=64, BK=32, 256 threads (8 waves). bf16 WMMA 16x16x32, f32 acc.
// ---------------------------------------------------------------------------
template<bool TRANSA, typename TA, bool DUAL>
__global__ __launch_bounds__(256)
void gemm64_kernel(const TA* __restrict__ A, long lda, int Kd,
                   const TA* __restrict__ A2, long lda2, int Kd2, int negB2,
                   const float* __restrict__ B, const float* __restrict__ B2,
                   const float* __restrict__ bias, const float* __restrict__ resid,
                   float* __restrict__ Cout, int M, int act)
{
  __shared__ bf16 sA[128][40];   // [m][k], 80B row stride (16B aligned)
  __shared__ bf16 sBt[64][40];   // [n][k], transposed B tile

  const int tid  = threadIdx.x;
  const int wave = tid >> 5;
  const int lane = tid & 31;
  const int m0   = blockIdx.x * 128;

  const int mA   = lane & 15;
  const int kbA  = (lane < 16) ? 0 : 8;
  const int kbB  = (lane < 16) ? 0 : 8;
  const int colB = lane & 15;

  const int rS  = tid >> 1;
  const int cS  = (tid & 1) * 16;
  const int nS  = tid & 63;
  const int kS  = (tid >> 6) * 8;

  v8f acc[4] = {};

  const int passes = DUAL ? 2 : 1;
  for (int pass = 0; pass < passes; ++pass) {
    const TA*    Ap   = pass ? A2   : A;
    const long   ldap = pass ? lda2 : lda;
    const float* Bp   = pass ? B2   : B;
    const int    Kp   = pass ? Kd2  : Kd;
    const bool   negB = pass ? (negB2 != 0) : false;

    int kBeg = 0, kEnd = Kp;
    if (gridDim.y > 1) {
      int chunk = (((Kp + (int)gridDim.y - 1) / (int)gridDim.y) + 31) & ~31;
      kBeg = (int)blockIdx.y * chunk;
      kEnd = kBeg + chunk;
      if (kEnd > Kp) kEnd = Kp;
    }

    for (int k0 = kBeg; k0 < kEnd; k0 += 32) {
      const bool fullK = (k0 + 32 <= kEnd);
      // ---- stage A tile: thread owns 16 contiguous K of one row ----
      if (!TRANSA) {
        int gm = m0 + rS;
        if (gm < M && fullK) {                         // fast vector path
          v8bfx p0, p1;
          load16(Ap + (long)gm * ldap + (k0 + cS), p0, p1);
          *(v8bfx*)&sA[rS][cS]     = p0;
          *(v8bfx*)&sA[rS][cS + 8] = p1;
        } else {                                       // guarded tail
          v8bfx p0, p1;
          #pragma unroll
          for (int j = 0; j < 8; ++j) {
            int gk0 = k0 + cS + j, gk1 = k0 + cS + 8 + j;
            p0[j] = (bf16)((gm < M && gk0 < kEnd) ? cvtf(Ap[(long)gm * ldap + gk0]) : 0.f);
            p1[j] = (bf16)((gm < M && gk1 < kEnd) ? cvtf(Ap[(long)gm * ldap + gk1]) : 0.f);
          }
          *(v8bfx*)&sA[rS][cS]     = p0;
          *(v8bfx*)&sA[rS][cS + 8] = p1;
        }
#if defined(__gfx1250__)
        if (k0 + 32 < kEnd && gm < M)
          __builtin_prefetch(Ap + (long)gm * ldap + (k0 + 32), 0, 1);
#endif
      } else {
        #pragma unroll
        for (int it = 0; it < 16; ++it) {
          int idx = tid + it * 256;
          int r = idx & 127, c = idx >> 7;
          int gm = m0 + r, gk = k0 + c;
          float v = 0.f;
          if (gm < M && gk < kEnd) v = cvtf(Ap[(long)gk * ldap + gm]);
          sA[r][c] = (bf16)v;
        }
      }
      // ---- stage B tile transposed; unguarded fast path when K-full ----
      {
        v8bfx pack;
        if (fullK) {
          const float* bp = Bp + (long)(k0 + kS) * 64 + nS;
          #pragma unroll
          for (int j = 0; j < 8; ++j) {
            float v = bp[(long)j * 64];
            pack[j] = (bf16)(negB ? -v : v);
          }
        } else {
          #pragma unroll
          for (int j = 0; j < 8; ++j) {
            int gk = k0 + kS + j;
            float v = (gk < kEnd) ? Bp[(long)gk * 64 + nS] : 0.f;
            pack[j] = (bf16)(negB ? -v : v);
          }
        }
        *(v8bfx*)&sBt[nS][kS] = pack;
      }
      __syncthreads();
      wmma_step(sA, sBt, wave, mA, kbA, kbB, colB, acc);
      __syncthreads();
    }
  }

  const int rowHalf = (lane < 16) ? 0 : 8;
  const bool split = (gridDim.y > 1);
  #pragma unroll
  for (int nt = 0; nt < 4; ++nt) {
    #pragma unroll
    for (int r = 0; r < 8; ++r) {
      int gm = m0 + wave * 16 + rowHalf + r;
      int gn = nt * 16 + colB;
      if (gm < M) {
        float v = acc[nt][r];
        long  o = (long)gm * 64 + gn;
        if (split) {
          atomicAdd(&Cout[o], v);
        } else {
          if (bias)  v += bias[gn];
          if (resid) v += resid[o];
          if (act == 1) v = fmaxf(v, 0.f);
          Cout[o] = v;
        }
      }
    }
  }
}

// ---------------------------------------------------------------------------
// bf16 GEMM with Tensor Data Mover A-tile staging (for rbf / rbf^T).
// Wave 0 issues TENSOR_LOAD_TO_LDS with a D# describing a 128x32 bf16 tile
// whose LDS destination is padded 4 DWORDs per 16 DWORDs -> exactly the
// sA[128][40] layout. TENSORcnt-synchronized, then workgroup barrier.
// ---------------------------------------------------------------------------
__global__ __launch_bounds__(256)
void gemm64_bf16_tdm(const bf16* __restrict__ A, long lda, int Kd,
                     const float* __restrict__ B, float* __restrict__ Cout, int M)
{
  __shared__ bf16 sA[128][40];
  __shared__ bf16 sBt[64][40];

  const int tid  = threadIdx.x;
  const int wave = tid >> 5;
  const int lane = tid & 31;
  const int m0   = blockIdx.x * 128;

  const int mA   = lane & 15;
  const int kbA  = (lane < 16) ? 0 : 8;
  const int kbB  = (lane < 16) ? 0 : 8;
  const int colB = lane & 15;
  const int nS   = tid & 63;
  const int kS   = (tid >> 6) * 8;

  const unsigned int ldsA = (unsigned int)(size_t)&sA[0][0];

  v8f acc[4] = {};

  for (int k0 = 0; k0 < Kd; k0 += 32) {
    if (wave == 0) {
      // ---- D# group 0: count=1 | lds_addr | global_addr(bytes) | type=2 ----
      unsigned long long ga =
          (unsigned long long)(size_t)(A + (long)m0 * lda + k0);
      u32x4 g0;
      g0[0] = 1u;
      g0[1] = ldsA;
      g0[2] = (unsigned int)ga;
      g0[3] = (unsigned int)(ga >> 32) | (2u << 30);
      // ---- D# group 1 ----
      unsigned int td0 = (unsigned int)(Kd - k0);   // remaining K (OOB -> 0)
      unsigned int td1 = (unsigned int)(M - m0);    // remaining M (OOB -> 0)
      unsigned long long st0 = (unsigned long long)lda;
      i32x8 g1;
      g1[0] = (int)((1u << 16)      // data_size = 2 bytes
                  | (1u << 20)      // pad_enable
                  | (3u << 22)      // pad_interval: every 16 DWORDs (=32 bf16 row)
                  | (3u << 25));    // pad_amount: 4 DWORDs (-> 80B row stride)
      g1[1] = (int)((td0 & 0xFFFFu) << 16);
      g1[2] = (int)((td0 >> 16) | ((td1 & 0xFFFFu) << 16));
      g1[3] = (int)((td1 >> 16) | (32u << 16));     // tile_dim0 = 32
      g1[4] = (int)128u;                            // tile_dim1 = 128, tile_dim2 = 0
      g1[5] = (int)(unsigned int)st0;
      g1[6] = (int)(unsigned int)(st0 >> 32);
      g1[7] = 0;
      i32x4 z4 = {};
      i32x8 z8 = {};
      __builtin_amdgcn_tensor_load_to_lds(g0, g1, z4, z4, z8, 0);
    }
    // ---- B tile transposed (f32 -> bf16), VALU path overlaps the TDM ----
    {
      v8bfx pack;
      if (k0 + 32 <= Kd) {
        const float* bp = B + (long)(k0 + kS) * 64 + nS;
        #pragma unroll
        for (int j = 0; j < 8; ++j) pack[j] = (bf16)bp[(long)j * 64];
      } else {
        #pragma unroll
        for (int j = 0; j < 8; ++j) {
          int gk = k0 + kS + j;
          pack[j] = (bf16)((gk < Kd) ? B[(long)gk * 64 + nS] : 0.f);
        }
      }
      *(v8bfx*)&sBt[nS][kS] = pack;
    }
    if (wave == 0) __builtin_amdgcn_s_wait_tensorcnt(0);
    __syncthreads();
    wmma_step(sA, sBt, wave, mA, kbA, kbB, colB, acc);
    __syncthreads();
  }

  const int rowHalf = (lane < 16) ? 0 : 8;
  #pragma unroll
  for (int nt = 0; nt < 4; ++nt) {
    #pragma unroll
    for (int r = 0; r < 8; ++r) {
      int gm = m0 + wave * 16 + rowHalf + r;
      if (gm < M) Cout[(long)gm * 64 + nt * 16 + colB] = acc[nt][r];
    }
  }
}

// ---------------------------------------------------------------------------
// Elementwise / graph kernels
// ---------------------------------------------------------------------------
__global__ void k_zero(float* p, int n) {
  int i = blockIdx.x * 256 + threadIdx.x;
  if (i < n) p[i] = 0.f;
}

__device__ __forceinline__ float rbf_val(const float* verts, const float* pos,
                                         int v, int n) {
  float dx = verts[v * 3 + 0] - pos[n * 3 + 0];
  float dy = verts[v * 3 + 1] - pos[n * 3 + 1];
  float dz = verts[v * 3 + 2] - pos[n * 3 + 2];
  float r  = sqrtf(dx * dx + dy * dy + dz * dz);
  return __expf(-r * 0.4f);                   // 1/SIGMA = 1/2.5
}

__global__ void k_rbf_vn(const float* __restrict__ verts, const float* __restrict__ pos,
                         bf16* __restrict__ out) {     // out[v][n], n fast
  int n = blockIdx.x * blockDim.x + threadIdx.x;
  int v = blockIdx.y * blockDim.y + threadIdx.y;
  if (n >= N_G || v >= V_N) return;
  out[(long)v * N_G + n] = (bf16)rbf_val(verts, pos, v, n);
}

__global__ void k_rbf_nv(const float* __restrict__ verts, const float* __restrict__ pos,
                         bf16* __restrict__ out) {     // out[n][v], v fast
  int v = blockIdx.x * blockDim.x + threadIdx.x;
  int n = blockIdx.y * blockDim.y + threadIdx.y;
  if (n >= N_G || v >= V_N) return;
  out[(long)n * V_N + v] = (bf16)rbf_val(verts, pos, v, n);
}

__global__ void k_mass(const float* __restrict__ x, const float* __restrict__ mass,
                       float* __restrict__ out, int n) {
  int i = blockIdx.x * 256 + threadIdx.x;
  if (i < n) out[i] = x[i] * mass[i >> 6];
}

__global__ void k_spec(const float* __restrict__ xs, const float* __restrict__ evals,
                       const float* __restrict__ tvec, float* __restrict__ out) {
  int i = blockIdx.x * 256 + threadIdx.x;
  if (i < KEV * 64) {
    int k = i >> 6, c = i & 63;
    float t = fmaxf(tvec[c], 1e-8f);
    out[i] = __expf(-evals[k] * t) * xs[i];
  }
}

__global__ void k_gfeat(const float* __restrict__ gr, const float* __restrict__ gi,
                        const float* __restrict__ br, const float* __restrict__ bi,
                        float* __restrict__ out, int n) {
  int i = blockIdx.x * 256 + threadIdx.x;
  if (i < n) out[i] = tanhf(gr[i] * br[i] + gi[i] * bi[i]);
}

__global__ void k_concat3(const float* __restrict__ a, const float* __restrict__ b,
                          const float* __restrict__ c, float* __restrict__ out, int n) {
  int i = blockIdx.x * 256 + threadIdx.x;
  if (i >= n) return;
  int v = i / 192, ch = i % 192;
  float val;
  if (ch < 64)        val = a[v * 64 + ch];
  else if (ch < 128)  val = b[v * 64 + (ch - 64)];
  else                val = c[v * 64 + (ch - 128)];
  out[i] = val;
}

__global__ void k_deg_init(float* deg, int n) {
  int i = blockIdx.x * 256 + threadIdx.x;
  if (i < n) deg[i] = 1.0f;
}

__global__ void k_deg_edges(const int* __restrict__ ei, const float* __restrict__ ew,
                            float* deg) {
  int e = blockIdx.x * 256 + threadIdx.x;
  if (e < E_N) atomicAdd(&deg[ei[E_N + e]], ew[e]);
}

__global__ void k_dinv(const float* __restrict__ deg, float* __restrict__ dinv, int n) {
  int i = blockIdx.x * 256 + threadIdx.x;
  if (i < n) { float d = deg[i]; dinv[i] = (d > 0.f) ? rsqrtf(d) : 0.f; }
}

__global__ void k_agg_init(const float* __restrict__ h, const float* __restrict__ dinv,
                           float* __restrict__ agg, int n) {
  int i = blockIdx.x * 256 + threadIdx.x;
  if (i < n) { float di = dinv[i >> 6]; agg[i] = h[i] * di * di; }
}

__global__ void k_agg_edges(const int* __restrict__ ei, const float* __restrict__ ew,
                            const float* __restrict__ dinv, const float* __restrict__ h,
                            float* __restrict__ agg) {
  long t = (long)blockIdx.x * 256 + threadIdx.x;
  if (t >= (long)E_N * 64) return;
  int e = (int)(t >> 6), c = (int)(t & 63);
  int s = ei[e], d = ei[E_N + e];
  float coef = dinv[s] * ew[e] * dinv[d];
  atomicAdd(&agg[(long)d * 64 + c], coef * h[(long)s * 64 + c]);
}

__global__ void k_fin(float* __restrict__ agg, const float* __restrict__ b,
                      int relu, int n) {
  int i = blockIdx.x * 256 + threadIdx.x;
  if (i < n) {
    float v = agg[i] + b[i & 63];
    if (relu) v = fmaxf(v, 0.f);
    agg[i] = v;
  }
}

// ---------------------------------------------------------------------------
static inline int ceil_div(int a, int b) { return (a + b - 1) / b; }

extern "C" void kernel_launch(void* const* d_in, const int* in_sizes, int n_in,
                              void* d_out, int out_size, void* d_ws, size_t ws_size,
                              hipStream_t stream) {
  (void)in_sizes; (void)n_in; (void)out_size; (void)ws_size;

  const float* x_in   = (const float*)d_in[0];
  const float* mass   = (const float*)d_in[1];
  const float* evals  = (const float*)d_in[2];
  const float* evecs  = (const float*)d_in[3];
  const float* gradX  = (const float*)d_in[4];
  const float* gradY  = (const float*)d_in[5];
  const float* verts  = (const float*)d_in[6];
  const float* graphx = (const float*)d_in[7];
  const float* gpos   = (const float*)d_in[8];
  const int*   eidx   = (const int*)d_in[9];
  const float* ew     = (const float*)d_in[10];
  const float* Wl1    = (const float*)d_in[11];
  const float* bl1    = (const float*)d_in[12];
  const float* Wl2    = (const float*)d_in[13];
  const float* bl2    = (const float*)d_in[14];
  const float* dtime  = (const float*)d_in[15];
  const float* Are    = (const float*)d_in[16];
  const float* Aim    = (const float*)d_in[17];
  const float* W0     = (const float*)d_in[18];
  const float* b0     = (const float*)d_in[19];
  const float* W1     = (const float*)d_in[20];
  const float* b1     = (const float*)d_in[21];
  const float* W2     = (const float*)d_in[22];
  const float* b2     = (const float*)d_in[23];
  const float* g1W    = (const float*)d_in[24];
  const float* g1b    = (const float*)d_in[25];
  const float* g2W    = (const float*)d_in[26];
  const float* g2b    = (const float*)d_in[27];
  const float* mixW   = (const float*)d_in[28];
  const float* mixb   = (const float*)d_in[29];
  const float* mlW    = (const float*)d_in[30];
  const float* mlb    = (const float*)d_in[31];

  uint8_t* wp = (uint8_t*)d_ws;
  auto alloc = [&](size_t bytes) -> void* {
    void* p = wp; wp += (bytes + 255) & ~(size_t)255; return p;
  };
  const size_t V64 = (size_t)V_N * 64 * sizeof(float);
  bf16*  rbfB  = (bf16*)alloc((size_t)V_N * N_G * sizeof(bf16));   // [v][n]
  bf16*  rbfT  = (bf16*)alloc((size_t)V_N * N_G * sizeof(bf16));   // [n][v]
  float* dA    = (float*)alloc(V64);   float* dB   = (float*)alloc(V64);
  float* gA    = (float*)alloc(V64);   float* gB   = (float*)alloc(V64);
  float* mx    = (float*)alloc(V64);   float* xdf  = (float*)alloc(V64);
  float* g_re  = (float*)alloc(V64);   float* g_im = (float*)alloc(V64);
  float* b_re  = (float*)alloc(V64);   float* b_im = (float*)alloc(V64);
  float* gfeat = (float*)alloc(V64);   float* h1   = (float*)alloc(V64);
  float* h2    = (float*)alloc(V64);   float* ddif = (float*)alloc(V64);
  float* dog   = (float*)alloc(V64);   float* god  = (float*)alloc(V64);
  float* hW    = (float*)alloc(V64);   float* aggA = (float*)alloc(V64);
  float* aggB  = (float*)alloc(V64);
  float* feat  = (float*)alloc((size_t)V_N * 192 * sizeof(float));
  float* xspec = (float*)alloc((size_t)KEV * 64 * sizeof(float));
  float* sbuf  = (float*)alloc((size_t)KEV * 64 * sizeof(float));
  float* deg   = (float*)alloc((size_t)N_G * sizeof(float));
  float* dinv  = (float*)alloc((size_t)N_G * sizeof(float));

  const float* FNULL = nullptr;

  auto gemmF = [&](const float* A, long lda, int Kd, const float* B,
                   const float* bias, const float* resid, float* C, int M, int act) {
    hipLaunchKernelGGL((gemm64_kernel<false, float, false>),
                       dim3(ceil_div(M, 128)), dim3(256), 0, stream,
                       A, lda, Kd, FNULL, (long)0, 0, 0, B, FNULL, bias, resid, C, M, act);
  };
  auto gemmDual = [&](const float* A, const float* B, const float* A2, const float* B2,
                      int neg, const float* bias, float* C, int M) {
    hipLaunchKernelGGL((gemm64_kernel<false, float, true>),
                       dim3(ceil_div(M, 128)), dim3(256), 0, stream,
                       A, (long)64, 64, A2, (long)64, 64, neg, B, B2, bias, FNULL, C, M, 0);
  };
  auto gemmBf = [&](const bf16* A, long lda, int Kd, const float* B,
                    float* C, int M) {
    hipLaunchKernelGGL(gemm64_bf16_tdm,
                       dim3(ceil_div(M, 128)), dim3(256), 0, stream,
                       A, lda, Kd, B, C, M);
  };

  const int nV64 = V_N * 64;
  dim3 B256(256);
  dim3 gE(ceil_div(nV64, 256));

  hipLaunchKernelGGL(k_rbf_vn, dim3(ceil_div(N_G, 64), ceil_div(V_N, 4)), dim3(64, 4),
                     0, stream, verts, gpos, rbfB);
  hipLaunchKernelGGL(k_rbf_nv, dim3(ceil_div(V_N, 64), ceil_div(N_G, 4)), dim3(64, 4),
                     0, stream, verts, gpos, rbfT);
  gemmF(x_in, 32, 32, Wl1, bl1, nullptr, dA, V_N, 0);       // diff_x0
  gemmF(graphx, 32, 32, Wl2, bl2, nullptr, gA, N_G, 0);     // gx0
  hipLaunchKernelGGL(k_deg_init, dim3(ceil_div(N_G, 256)), B256, 0, stream, deg, N_G);
  hipLaunchKernelGGL(k_deg_edges, dim3(ceil_div(E_N, 256)), B256, 0, stream, eidx, ew, deg);
  hipLaunchKernelGGL(k_dinv, dim3(ceil_div(N_G, 256)), B256, 0, stream, deg, dinv, N_G);

  float* dcur = dA; float* dnxt = dB;
  float* gcur = gA; float* gnxt = gB;

  for (int i = 0; i < 4; ++i) {
    // ============ diffusion block ============
    hipLaunchKernelGGL(k_mass, gE, B256, 0, stream, dcur, mass, mx, nV64);
    hipLaunchKernelGGL(k_zero, dim3(32), B256, 0, stream, xspec, KEV * 64);
    hipLaunchKernelGGL((gemm64_kernel<true, float, false>),     // x_spec = evecs^T @ mx
                       dim3(1, 32), dim3(256), 0, stream,
                       evecs, (long)KEV, V_N, FNULL, (long)0, 0, 0,
                       mx, FNULL, FNULL, FNULL, xspec, KEV, 0);
    hipLaunchKernelGGL(k_spec, dim3(32), B256, 0, stream, xspec, evals, dtime + i * 64, sbuf);
    gemmF(evecs, KEV, KEV, sbuf, nullptr, nullptr, xdf, V_N, 0);        // x_diff
    gemmF(gradX, N_G, V_N, xdf, nullptr, nullptr, g_re, V_N, 0);        // g_re
    gemmF(gradY, N_G, V_N, xdf, nullptr, nullptr, g_im, V_N, 0);        // g_im
    gemmDual(g_re, Are + i * 4096, g_im, Aim + i * 4096, 1, nullptr, b_re, V_N);
    gemmDual(g_im, Are + i * 4096, g_re, Aim + i * 4096, 0, nullptr, b_im, V_N);
    hipLaunchKernelGGL(k_gfeat, gE, B256, 0, stream, g_re, g_im, b_re, b_im, gfeat, nV64);
    hipLaunchKernelGGL(k_concat3, dim3(ceil_div(V_N * 192, 256)), B256, 0, stream,
                       dcur, xdf, gfeat, feat, V_N * 192);
    gemmF(feat, 192, 192, W0 + i * 192 * 64, b0 + i * 64, nullptr, h1, V_N, 1);
    gemmF(h1, 64, 64, W1 + i * 4096, b1 + i * 64, nullptr, h2, V_N, 1);
    gemmF(h2, 64, 64, W2 + i * 4096, b2 + i * 64, dcur, ddif, V_N, 0);  // + residual

    // ============ GCN (two convs) ============
    gemmF(gcur, 64, 64, g1W + i * 4096, nullptr, nullptr, hW, N_G, 0);
    hipLaunchKernelGGL(k_agg_init, gE, B256, 0, stream, hW, dinv, aggA, nV64);
    hipLaunchKernelGGL(k_agg_edges, dim3(ceil_div(E_N * 64, 256)), B256, 0, stream,
                       eidx, ew, dinv, hW, aggA);
    hipLaunchKernelGGL(k_fin, gE, B256, 0, stream, aggA, g1b + i * 64, 1, nV64);
    gemmF(aggA, 64, 64, g2W + i * 4096, nullptr, nullptr, hW, N_G, 0);
    hipLaunchKernelGGL(k_agg_init, gE, B256, 0, stream, hW, dinv, aggB, nV64);
    hipLaunchKernelGGL(k_agg_edges, dim3(ceil_div(E_N * 64, 256)), B256, 0, stream,
                       eidx, ew, dinv, hW, aggB);
    hipLaunchKernelGGL(k_fin, gE, B256, 0, stream, aggB, g2b + i * 64, 0, nV64);

    // ============ RBF cross-domain (bf16 WMMA + TDM A-tiles) ============
    gemmBf(rbfT, V_N, V_N, ddif, dog, N_G);                  // dog = rbf^T @ ddif
    gemmBf(rbfB, N_G, N_G, aggB, god, V_N);                  // god = rbf  @ gx

    // ============ mixers (concat-GEMM as dual GEMM) ============
    const float* Wm = (i < 3) ? (mixW + i * 128 * 64) : mlW;
    const float* bm = (i < 3) ? (mixb + i * 64)       : mlb;
    float* dOutBuf  = (i == 3) ? (float*)d_out : dnxt;
    if (i < 3)
      gemmDual(dog, Wm, aggB, Wm + 64 * 64, 0, bm, gnxt, N_G);   // gx_next
    gemmDual(ddif, Wm, god, Wm + 64 * 64, 0, bm, dOutBuf, V_N);  // diff_x_next

    float* t = dcur; dcur = (i == 3) ? dcur : dnxt; dnxt = t;
    t = gcur; gcur = gnxt; gnxt = t;
  }
}
